// ConvLEM_27238682591836
// MI455X (gfx1250) — matrix operand order repaired and checked by reading, not compiled, not run on hardware
//
#include <hip/hip_runtime.h>

typedef __attribute__((ext_vector_type(16))) _Float16 v16h;
typedef __attribute__((ext_vector_type(8)))  _Float16 v8h;
typedef __attribute__((ext_vector_type(8)))  float    v8f;

#define WMMA16(A,B,C) __builtin_amdgcn_wmma_f32_16x16x32_f16(false,(A),false,(B),(short)0,(C),false,false)

// Problem constants: B=8, Cin=Cout=32, T=50, H=W=64, k=3, SAME pad.
// Spatial tile per block: 8 rows x 16 cols; 8 waves, one row (M=16 pixels) each.
// LDS patch: 10 x 18 cells, 32 f16 channels per cell (channel-innermost).

__device__ __forceinline__ v16h ldsA(const _Float16* s, int cell, int koff) {
    // A 16x32 f16 fragment: lane half-group needs cin {koff..koff+7, koff+16..koff+23}
    const v8h a0 = *(const v8h*)(s + cell * 32 + koff);
    const v8h a1 = *(const v8h*)(s + cell * 32 + koff + 16);
    return __builtin_shufflevector(a0, a1, 0,1,2,3,4,5,6,7,8,9,10,11,12,13,14,15);
}

__device__ __forceinline__ v16h ldB(const _Float16* w, int tile, int pos, int lane) {
    // Pre-packed B 32x16 fragment: 16 contiguous halves per lane (32B load).
    return *(const v16h*)(w + (size_t)(((tile * 9 + pos) * 32 + lane) * 16));
}

__device__ __forceinline__ float sigmoidf_(float x) { return 1.f / (1.f + __expf(-x)); }

// ---------------------------------------------------------------------------
// Weight repack: W[co][cin][kh][kw] f32 -> f16 B-fragments
// frag addr = ((tile*9 + pos)*32 + lane)*16 + j ; co = tile*16 + (lane&15),
// cin = (lane>=16 ? 8:0) + (j&7) + (j>=8 ? 16:0), pos = kh*3+kw.
// ---------------------------------------------------------------------------
__global__ __launch_bounds__(256) void lem_prep_w(
    const float* __restrict__ Wx, const float* __restrict__ Wy,
    const float* __restrict__ Wz,
    _Float16* __restrict__ wxF, _Float16* __restrict__ wyF,
    _Float16* __restrict__ wzF)
{
    int idx = blockIdx.x * 256 + threadIdx.x;
    int j    = idx & 15;
    int lane = (idx >> 4) & 31;
    int tp   = idx >> 9;
    int pos  = tp % 9;
    int tile = tp / 9;
    int n  = lane & 15;
    int k  = ((lane >> 4) << 3) + (j & 7) + ((j >> 3) << 4);
    int kh = pos / 3, kw = pos - kh * 3;
    size_t src = ((size_t)((tile * 16 + n) * 32 + k) * 3 + kh) * 3 + kw;
    if (idx < 8 * 9 * 512) wxF[idx] = (_Float16)Wx[src];
    if (idx < 6 * 9 * 512) wyF[idx] = (_Float16)Wy[src];
    if (idx < 2 * 9 * 512) wzF[idx] = (_Float16)Wz[src];
}

__global__ __launch_bounds__(256) void lem_zero(float* __restrict__ p, int n) {
    int i = blockIdx.x * 256 + threadIdx.x;
    if (i < n) p[i] = 0.f;
}

// ---------------------------------------------------------------------------
// Kernel A (per step): conv(x_t,Wx) + conv(Y,Wy) fused with gate pointwise.
// Updates Z in place; writes ms2-logit and xy-term to workspace.
// ---------------------------------------------------------------------------
__global__ __launch_bounds__(256) void lem_gates(
    const float* __restrict__ X,
    const float* __restrict__ bx, const float* __restrict__ by,
    const _Float16* __restrict__ wxF, const _Float16* __restrict__ wyF,
    const float* __restrict__ Yst, float* __restrict__ Zst,
    float* __restrict__ g2l, float* __restrict__ xyb, int t)
{
    __shared__ __align__(16) _Float16 sX[5760];   // 10*18*32
    __shared__ __align__(16) _Float16 sY[5760];

    const int b  = blockIdx.x >> 5;
    const int th = (blockIdx.x >> 2) & 7;
    const int tw = blockIdx.x & 3;
    const int h0 = th * 8, w0 = tw * 16;
    const int tid = threadIdx.x;

    for (int i = tid; i < 5760; i += 256) {
        int cin = i & 31, cell = i >> 5;
        int ph = cell / 18, pw = cell - ph * 18;
        int h = h0 + ph - 1, w = w0 + pw - 1;
        float xv = 0.f, yv = 0.f;
        if (h >= 0 && h < 64 && w >= 0 && w < 64) {
            xv = X[((size_t)((b * 32 + cin) * 50 + t)) * 4096 + h * 64 + w];
            yv = Yst[((size_t)(b * 32 + cin)) * 4096 + h * 64 + w];
        }
        sX[cell * 32 + cin] = (_Float16)xv;
        sY[cell * 32 + cin] = (_Float16)yv;
    }
    __syncthreads();

    const int wv = tid >> 5, lane = tid & 31;
    const int wl = lane & 15;
    const int koff = (lane >> 4) << 3;

    for (int ct = 0; ct < 2; ++ct) {
        v8f axg1 = {}, axg2 = {}, axz = {}, axy = {};
        v8f ayg1 = {}, ayg2 = {}, ayz = {};
        #pragma unroll
        for (int pos = 0; pos < 9; ++pos) {
            const int kh = pos / 3, kw = pos - kh * 3;
            const int cell = (wv + kh) * 18 + (wl + kw);
            v16h ax = ldsA(sX, cell, koff);
            v16h ay = ldsA(sY, cell, koff);
            axg1 = WMMA16(ax, ldB(wxF, 0 + ct, pos, lane), axg1);
            axg2 = WMMA16(ax, ldB(wxF, 2 + ct, pos, lane), axg2);
            axz  = WMMA16(ax, ldB(wxF, 4 + ct, pos, lane), axz);
            axy  = WMMA16(ax, ldB(wxF, 6 + ct, pos, lane), axy);
            ayg1 = WMMA16(ay, ldB(wyF, 0 + ct, pos, lane), ayg1);
            ayg2 = WMMA16(ay, ldB(wyF, 2 + ct, pos, lane), ayg2);
            ayz  = WMMA16(ay, ldB(wyF, 4 + ct, pos, lane), ayz);
        }
        // D layout: lane l, reg r -> pixel m = r + 8*(l>>4), channel n = l&15.
        const int c = ct * 16 + wl;
        const float bx1 = bx[c],      by1 = by[c];
        const float bx2 = bx[32 + c], by2 = by[32 + c];
        const float bx3 = bx[64 + c], by3 = by[64 + c];
        const float bx4 = bx[96 + c];
        const int mh = (lane >> 4) << 3;
        const int h = h0 + wv;
        #pragma unroll
        for (int r = 0; r < 8; ++r) {
            const int w = w0 + mh + r;
            const size_t idx = ((size_t)(b * 32 + c)) * 4096 + h * 64 + w;
            float ms1 = sigmoidf_(axg1[r] + ayg1[r] + bx1 + by1);   // DT = 1
            float zc  = tanhf(axz[r] + ayz[r] + bx3 + by3);
            float zn  = (1.f - ms1) * Zst[idx] + ms1 * zc;
            Zst[idx] = zn;
            g2l[idx] = axg2[r] + ayg2[r] + bx2 + by2;
            xyb[idx] = axy[r] + bx4;
        }
    }
}

// ---------------------------------------------------------------------------
// Kernel B (per step): conv(Z_new,Wz) fused with Y update; writes output[:, :, t].
// ---------------------------------------------------------------------------
__global__ __launch_bounds__(256) void lem_update(
    const float* __restrict__ Zst, const _Float16* __restrict__ wzF,
    const float* __restrict__ bz,
    const float* __restrict__ g2l, const float* __restrict__ xyb,
    float* __restrict__ Yst, float* __restrict__ out, int t)
{
    __shared__ __align__(16) _Float16 sZ[5760];

    const int b  = blockIdx.x >> 5;
    const int th = (blockIdx.x >> 2) & 7;
    const int tw = blockIdx.x & 3;
    const int h0 = th * 8, w0 = tw * 16;
    const int tid = threadIdx.x;

    for (int i = tid; i < 5760; i += 256) {
        int cin = i & 31, cell = i >> 5;
        int ph = cell / 18, pw = cell - ph * 18;
        int h = h0 + ph - 1, w = w0 + pw - 1;
        float zv = 0.f;
        if (h >= 0 && h < 64 && w >= 0 && w < 64)
            zv = Zst[((size_t)(b * 32 + cin)) * 4096 + h * 64 + w];
        sZ[cell * 32 + cin] = (_Float16)zv;
    }
    __syncthreads();

    const int wv = tid >> 5, lane = tid & 31;
    const int wl = lane & 15;
    const int koff = (lane >> 4) << 3;

    for (int ct = 0; ct < 2; ++ct) {
        v8f acc = {};
        #pragma unroll
        for (int pos = 0; pos < 9; ++pos) {
            const int kh = pos / 3, kw = pos - kh * 3;
            const int cell = (wv + kh) * 18 + (wl + kw);
            acc = WMMA16(ldsA(sZ, cell, koff), ldB(wzF, ct, pos, lane), acc);
        }
        const int c = ct * 16 + wl;
        const float bzc = bz[c];
        const int mh = (lane >> 4) << 3;
        const int h = h0 + wv;
        #pragma unroll
        for (int r = 0; r < 8; ++r) {
            const int w = w0 + mh + r;
            const size_t idx = ((size_t)(b * 32 + c)) * 4096 + h * 64 + w;
            float ms2 = sigmoidf_(g2l[idx]);                         // DT = 1
            float yc  = tanhf(xyb[idx] + acc[r] + bzc);
            float yn  = (1.f - ms2) * Yst[idx] + ms2 * yc;
            Yst[idx] = yn;
            out[((size_t)((b * 32 + c) * 50 + t)) * 4096 + h * 64 + w] = yn;
        }
    }
}

// ---------------------------------------------------------------------------
extern "C" void kernel_launch(void* const* d_in, const int* in_sizes, int n_in,
                              void* d_out, int out_size, void* d_ws, size_t ws_size,
                              hipStream_t stream)
{
    const float* X  = (const float*)d_in[0];
    const float* Wx = (const float*)d_in[1];
    const float* bx = (const float*)d_in[2];
    const float* Wy = (const float*)d_in[3];
    const float* by = (const float*)d_in[4];
    const float* Wz = (const float*)d_in[5];
    const float* bz = (const float*)d_in[6];
    float* out = (float*)d_out;

    const size_t NST = (size_t)8 * 32 * 64 * 64;   // 1,048,576 floats per state
    float* Yst = (float*)d_ws;
    float* Zst = Yst + NST;
    float* g2l = Zst + NST;
    float* xyb = g2l + NST;
    _Float16* wxF = (_Float16*)(xyb + NST);        // 8 tiles * 9 pos * 512
    _Float16* wyF = wxF + 8 * 9 * 512;             // 6 tiles
    _Float16* wzF = wyF + 6 * 9 * 512;             // 2 tiles

    lem_zero<<<(int)((2 * NST + 255) / 256), 256, 0, stream>>>(Yst, (int)(2 * NST));
    lem_prep_w<<<144, 256, 0, stream>>>(Wx, Wy, Wz, wxF, wyF, wzF);

    for (int t = 0; t < 50; ++t) {
        lem_gates<<<256, 256, 0, stream>>>(X, bx, by, wxF, wyF, Yst, Zst, g2l, xyb, t);
        lem_update<<<256, 256, 0, stream>>>(Zst, wzF, bz, g2l, xyb, Yst, out, t);
    }
}
